// PrimitiveDecoderKnownLength_49675591746070
// MI455X (gfx1250) — matrix operand order repaired and checked by reading, not compiled, not run on hardware
//
#include <hip/hip_runtime.h>
#include <hip/hip_bf16.h>
#include <stdint.h>

#define NH      2048
#define GATES   8192      // 4*NH
#define T_STEPS 256
#define NZ      256
#define NS      64
#define RB      128       // blocks in persistent recurrent kernel
#define HPB     16        // hidden units per block (NH/RB)

typedef __attribute__((ext_vector_type(16))) __bf16 v16bf;
typedef __attribute__((ext_vector_type(8)))  float  v8f;

__device__ __forceinline__ unsigned short f2bf(float f) {
  unsigned u = __builtin_bit_cast(unsigned, f);
  unsigned r = u + 0x7FFFu + ((u >> 16) & 1u);   // round-to-nearest-even
  return (unsigned short)(r >> 16);
}
__device__ __forceinline__ float bf_lo(unsigned p) { return __builtin_bit_cast(float, p << 16); }
__device__ __forceinline__ float bf_hi(unsigned p) { return __builtin_bit_cast(float, p & 0xFFFF0000u); }
__device__ __forceinline__ float sigmoidf_(float x) { return 1.0f / (1.0f + __expf(-x)); }

// ---------------- small utility kernels ----------------

__global__ void k_reset(float* h, unsigned* bar) {
  int i = blockIdx.x * blockDim.x + threadIdx.x;
  if (i < NH) h[i] = 0.f;
  if (i < 2)  bar[i] = 0u;
}

__global__ void k_zero_f32(float* p, int n) {
  int i = blockIdx.x * blockDim.x + threadIdx.x;
  int stride = gridDim.x * blockDim.x;
  for (; i < n; i += stride) p[i] = 0.f;
}

__global__ void k_bias(const float* __restrict__ bih0, const float* __restrict__ bhh0,
                       const float* __restrict__ bihr, const float* __restrict__ bhhr,
                       float* __restrict__ out) {
  int i = blockIdx.x * blockDim.x + threadIdx.x;           // 0 .. 4*GATES-1
  if (i >= 4 * GATES) return;
  int l = i >> 13, r = i & (GATES - 1);
  out[i] = (l == 0) ? (bih0[r] + bhh0[r])
                    : (bihr[(l - 1) * GATES + r] + bhhr[(l - 1) * GATES + r]);
}

__global__ void k_f32_to_bf16(const float* __restrict__ src, unsigned short* __restrict__ dst, int n) {
  int i = blockIdx.x * blockDim.x + threadIdx.x;
  int stride = gridDim.x * blockDim.x;
  for (; i < n; i += stride) dst[i] = f2bf(src[i]);
}

// layer-0 input contribution: preGates[0, r] = z . w_ih0[r, :]   (t>0 rows stay zero)
__global__ void k_gemv_in0(const float* __restrict__ z, const float* __restrict__ w,
                           float* __restrict__ pre) {
  int gw = (blockIdx.x * blockDim.x + threadIdx.x) >> 5;
  int lane = threadIdx.x & 31;
  if (gw >= GATES) return;
  const float* row = w + (size_t)gw * NZ;
  float acc = 0.f;
  for (int k = lane; k < NZ; k += 32) acc = __builtin_fmaf(row[k], z[k], acc);
  for (int o = 16; o; o >>= 1) acc += __shfl_xor(acc, o, 32);
  if (lane == 0) pre[gw] = acc;
}

// ---------------- WMMA bf16 GEMM: C[M,N] = A[M,K] * W[N,K]^T ----------------
// A row-major bf16 [M,K]; W row-major bf16 [N,K]; C fp32 [M,N].
// One wave owns a 16x16 D tile; block = 8 waves = 32(M) x 64(N).
__global__ void __launch_bounds__(256) k_gemm_wmma(const unsigned short* __restrict__ A,
                                                   const unsigned short* __restrict__ W,
                                                   float* __restrict__ C,
                                                   int M, int N, int K) {
  const int wave = threadIdx.x >> 5;
  const int lane = threadIdx.x & 31;
  const int wm = wave >> 2, wn = wave & 3;
  const int tileM = blockIdx.y * 32 + wm * 16;
  const int tileN = blockIdx.x * 64 + wn * 16;
  const int lo = lane & 15, hi = lane >> 4;

  union Frag { v16bf v; unsigned u[8]; uint4 q[2]; };
  Frag a, b;
  v8f acc = {};

  const int m = tileM + lo;
  const int n = tileN + lo;
  const uint4* arow = (const uint4*)(A + (size_t)m * K);   // 8 bf16 per uint4
  const uint4* brow = (const uint4*)(W + (size_t)n * K);

  for (int kk = 0; kk < K; kk += 32) {
    // A 16x32 bf16: lanes 0-15 hold K=[kk..kk+7] & [kk+16..kk+23]; lanes 16-31 shift by 8.
    int ka = (kk >> 3) + hi;           // uint4 index
    a.q[0] = arow[ka];                 // K = kk + hi*8 .. +7
    a.q[1] = arow[ka + 2];             // K = kk + 16 + hi*8 .. +7
    // B 32x16 bf16: lane holds column n, K = kk + hi*8 .. +7 & kk + 16 + hi*8 .. +7 (same as A)
    int kb = (kk >> 3) + hi;           // uint4 index
    b.q[0] = brow[kb];
    b.q[1] = brow[kb + 2];

    acc = __builtin_amdgcn_wmma_f32_16x16x32_bf16(false, a.v, false, b.v,
                                                  (short)0, acc, false, false);
  }
  // D layout: VGPR r -> M = tileM + r + 8*hi, N = tileN + lo
#pragma unroll
  for (int r = 0; r < 8; ++r) {
    int mm = tileM + r + 8 * hi;
    C[(size_t)mm * N + n] = acc[r];
  }
}

// ---------------- persistent recurrent LSTM layer ----------------
// 128 blocks x 256 threads; block owns 16 hidden units; grid barrier between steps.
// W_hh is bf16 (half the L2 traffic of fp32); accumulation in fp32.
__global__ void __launch_bounds__(256) k_lstm_seq(const unsigned short* __restrict__ w_hh, // [GATES, NH] bf16
                                                  const float* __restrict__ pre,    // [T, GATES] fp32
                                                  const float* __restrict__ bias,   // [GATES] fused
                                                  float* h_buf,                     // [NH] global h
                                                  float* __restrict__ hs_out,       // [T, NH] fp32
                                                  unsigned short* __restrict__ hs_bf, // [T, NH] bf16
                                                  unsigned* bar) {
  __shared__ float h_sh[NH];
  __shared__ float g_sh[64];
  __shared__ float c_sh[HPB];
  const int tid = threadIdx.x, wave = tid >> 5, lane = tid & 31;
  const int j0 = blockIdx.x * HPB;
  unsigned* cnt = bar;
  unsigned* gen = bar + 1;

  if (tid < HPB) c_sh[tid] = 0.f;

  for (int t = 0; t < T_STEPS; ++t) {
    __syncthreads();
    for (int i = tid; i < NH; i += 256) h_sh[i] = h_buf[i];
    __syncthreads();

    // 64 gate rows per block (4 gates x 16 hidden), 8 per wave
    for (int q = wave; q < 64; q += 8) {
      int r = (q >> 4) * NH + j0 + (q & 15);
      const uint2* wrow = (const uint2*)(w_hh + (size_t)r * NH);  // 4 bf16 per uint2
      float acc = 0.f;
      for (int i = lane; i < NH / 4; i += 32) {
        uint2 p = wrow[i];
        int k = i * 4;
        acc = __builtin_fmaf(bf_lo(p.x), h_sh[k + 0], acc);
        acc = __builtin_fmaf(bf_hi(p.x), h_sh[k + 1], acc);
        acc = __builtin_fmaf(bf_lo(p.y), h_sh[k + 2], acc);
        acc = __builtin_fmaf(bf_hi(p.y), h_sh[k + 3], acc);
      }
      for (int o = 16; o; o >>= 1) acc += __shfl_xor(acc, o, 32);
      if (lane == 0) g_sh[q] = acc + pre[t * GATES + r] + bias[r];
    }
    __syncthreads();

    if (tid < HPB) {
      int j = tid;
      float gi = g_sh[j], gf = g_sh[16 + j], gg = g_sh[32 + j], go = g_sh[48 + j];
      float iv = sigmoidf_(gi), fv = sigmoidf_(gf), ov = sigmoidf_(go), gv = tanhf(gg);
      float c = fv * c_sh[j] + iv * gv;
      c_sh[j] = c;
      float h = ov * tanhf(c);
      h_buf[j0 + j] = h;
      hs_out[t * NH + j0 + j] = h;
      hs_bf[t * NH + j0 + j] = f2bf(h);
      __threadfence();   // make h visible device-wide before barrier arrive
    }
    __syncthreads();

    // sense-reversal grid barrier (all RB blocks resident)
    if (tid == 0) {
      unsigned prev = __hip_atomic_fetch_add(cnt, 1u, __ATOMIC_ACQ_REL, __HIP_MEMORY_SCOPE_AGENT);
      if (prev == RB - 1) {
        __hip_atomic_store(cnt, 0u, __ATOMIC_RELAXED, __HIP_MEMORY_SCOPE_AGENT);
        __hip_atomic_fetch_add(gen, 1u, __ATOMIC_RELEASE, __HIP_MEMORY_SCOPE_AGENT);
      } else {
        unsigned target = (unsigned)(t + 1);
        while (__hip_atomic_load(gen, __ATOMIC_ACQUIRE, __HIP_MEMORY_SCOPE_AGENT) < target)
          __builtin_amdgcn_s_sleep(2);
      }
    }
    __syncthreads();
  }
}

// ---------------- final projection: out[t,s] = hs[t,:] . w_sp[s,:] + b_sp[s] ----------------
__global__ void k_proj(const float* __restrict__ hs, const float* __restrict__ wsp,
                       const float* __restrict__ bsp, float* __restrict__ out) {
  int gw = (blockIdx.x * blockDim.x + threadIdx.x) >> 5;   // 0 .. T*NS-1
  int lane = threadIdx.x & 31;
  if (gw >= T_STEPS * NS) return;
  int t = gw >> 6, s = gw & 63;
  const float* hrow = hs + (size_t)t * NH;
  const float* wrow = wsp + (size_t)s * NH;
  float acc = 0.f;
  for (int k = lane; k < NH; k += 32) acc = __builtin_fmaf(hrow[k], wrow[k], acc);
  for (int o = 16; o; o >>= 1) acc += __shfl_xor(acc, o, 32);
  if (lane == 0) out[t * NS + s] = acc + bsp[s];
}

// ---------------- host-side orchestration ----------------
extern "C" void kernel_launch(void* const* d_in, const int* in_sizes, int n_in,
                              void* d_out, int out_size, void* d_ws, size_t ws_size,
                              hipStream_t stream) {
  const float* z      = (const float*)d_in[0];
  const float* w_ih0  = (const float*)d_in[1];
  const float* w_hh0  = (const float*)d_in[2];
  const float* b_ih0  = (const float*)d_in[3];
  const float* b_hh0  = (const float*)d_in[4];
  const float* w_ih_r = (const float*)d_in[5];
  const float* w_hh_r = (const float*)d_in[6];
  const float* b_ih_r = (const float*)d_in[7];
  const float* b_hh_r = (const float*)d_in[8];
  const float* w_sp   = (const float*)d_in[9];
  const float* b_sp   = (const float*)d_in[10];

  char* ws = (char*)d_ws;
  unsigned*       bar     = (unsigned*)(ws + 0);                 // 2 u32
  float*          h_buf   = (float*)(ws + 1024);                 // NH f32
  float*          biasF   = (float*)(ws + 16384);                // 4*GATES f32
  float*          hs_f32  = (float*)(ws + (256u << 10));         // T*NH f32  (2 MB)
  unsigned short* hs_bf   = (unsigned short*)(ws + (4u << 20));  // T*NH bf16 (1 MB)
  float*          pre     = (float*)(ws + (8u << 20));           // T*GATES f32 (8 MB)
  unsigned short* wih_bf  = (unsigned short*)(ws + (16u << 20)); // GATES*NH bf16 (32 MB)
  unsigned short* whh_bf  = (unsigned short*)(ws + (48u << 20)); // GATES*NH bf16 (32 MB)
  float*          outp    = (float*)d_out;

  const int WN = GATES * NH;  // 16,777,216 weights per matrix

  // fused biases for all 4 layers
  k_bias<<<(4 * GATES + 255) / 256, 256, 0, stream>>>(b_ih0, b_hh0, b_ih_r, b_hh_r, biasF);

  // layer 0 input contribution: zero preGates, fill t=0 row via GEMV
  k_zero_f32<<<2048, 256, 0, stream>>>(pre, T_STEPS * GATES);
  k_gemv_in0<<<GATES * 32 / 256, 256, 0, stream>>>(z, w_ih0, pre);

  // layer 0 recurrent sweep (bf16 W_hh stays L2-resident across 256 steps)
  k_f32_to_bf16<<<4096, 256, 0, stream>>>(w_hh0, whh_bf, WN);
  k_reset<<<(NH + 255) / 256, 256, 0, stream>>>(h_buf, bar);
  k_lstm_seq<<<RB, 256, 0, stream>>>(whh_bf, pre, biasF, h_buf, hs_f32, hs_bf, bar);

  for (int l = 0; l < 3; ++l) {
    // batched input GEMM for next layer: preGates = hs_bf16 @ W_ih^T  (WMMA bf16)
    k_f32_to_bf16<<<4096, 256, 0, stream>>>(w_ih_r + (size_t)l * WN, wih_bf, WN);
    dim3 g(GATES / 64, T_STEPS / 32);
    k_gemm_wmma<<<g, 256, 0, stream>>>(hs_bf, wih_bf, pre, T_STEPS, GATES, NH);

    // recurrent sweep for layer l+1
    k_f32_to_bf16<<<4096, 256, 0, stream>>>(w_hh_r + (size_t)l * WN, whh_bf, WN);
    k_reset<<<(NH + 255) / 256, 256, 0, stream>>>(h_buf, bar);
    k_lstm_seq<<<RB, 256, 0, stream>>>(whh_bf, pre, biasF + (size_t)(l + 1) * GATES,
                                       h_buf, hs_f32, hs_bf, bar);
  }

  // output projection (fp32, tiny)
  k_proj<<<T_STEPS * NS * 32 / 256, 256, 0, stream>>>(hs_f32, w_sp, b_sp, outp);
}
